// SA_Block_43576738185508
// MI455X (gfx1250) — matrix operand order repaired and checked by reading, not compiled, not run on hardware
//
#include <hip/hip_runtime.h>

// ---------------------------------------------------------------------------
// SA_Block attention, CDNA5 (gfx1250) WMMA implementation.
// B=64, C=64, L=256, D(QK=V)=32, S = n_strokes (15).
// Grid: (B, 2) blocks of 256 threads (8 wave32 waves).
// ---------------------------------------------------------------------------

typedef __attribute__((ext_vector_type(16))) _Float16 v16h;
typedef __attribute__((ext_vector_type(8)))  _Float16 v8h;
typedef __attribute__((ext_vector_type(8)))  float    v8f;

#define Cc 64
#define Ll 256
#define Dd 32
#define INV_SCALE 0.17677669529663687f   // 1/sqrt(32)

__device__ __forceinline__ v8f wmma16(v16h a, v16h b, v8f c) {
  // D = A(16x32 f16) * B(32x16 f16) + C(16x16 f32)
  return __builtin_amdgcn_wmma_f32_16x16x32_f16(
      /*neg_a=*/false, a, /*neg_b=*/false, b,
      /*c_mod=*/(short)0, c, /*reuse_a=*/false, /*reuse_b=*/false);
}

__device__ __forceinline__ v16h cat16(v8h lo, v8h hi) {
  return __builtin_shufflevector(lo, hi, 0, 1, 2, 3, 4, 5, 6, 7,
                                         8, 9, 10, 11, 12, 13, 14, 15);
}

// A-matrix fragment (16x32 f16, M x K): lane = row (lane&15),
// halves 0..15 = K(16*hs + 0..15).
__device__ __forceinline__ v16h load_a_frag(const _Float16* base, int stride) {
  const int lane = threadIdx.x & 31;
  const int m = lane & 15, hs = lane >> 4;
  const _Float16* p = base + m * stride + hs * 16;
  v8h lo = *(const v8h*)(p);
  v8h hi = *(const v8h*)(p + 8);
  return cat16(lo, hi);
}

// B-matrix fragment (32x16 f16, K x N): lane = col (lane&15),
// halves 0..15 = K(16*hs + 0..15). base points at (K=0, N=0), K contiguous.
__device__ __forceinline__ v16h load_b_frag(const _Float16* base, int stride) {
  const int lane = threadIdx.x & 31;
  const int n = lane & 15, hs = lane >> 4;
  const _Float16* p = base + n * stride + hs * 16;
  v8h lo = *(const v8h*)(p);
  v8h hi = *(const v8h*)(p + 8);
  return cat16(lo, hi);
}

// Weight B-fragment straight from global f32 w[32][64] (B[k=c][n=d] = w[d][c]).
__device__ __forceinline__ v16h load_w_frag(const float* wmat, int nt, int ks) {
  const int lane = threadIdx.x & 31;
  const int n = lane & 15, hs = lane >> 4;
  const float* p = wmat + (nt * 16 + n) * Cc + ks * 32 + hs * 16;
  v16h r;
#pragma unroll
  for (int i = 0; i < 16; ++i) r[i] = (_Float16)p[i];
  return r;
}

__global__ __launch_bounds__(256)
void sa_block_wmma(const float* __restrict__ x,
                   const float* __restrict__ wq,
                   const float* __restrict__ wk,
                   const float* __restrict__ wv,
                   const float* __restrict__ bvp,
                   const int*   __restrict__ sidx,
                   const int*   __restrict__ nsp,
                   float*       __restrict__ out) {
  __shared__ _Float16 smem[Ll * Cc];        // 32KB: x^T staging, then K|V buffers
  __shared__ _Float16 pbuf[8 * 16 * 32];    // 8KB: per-wave Q/P transpose bounce

  const int b    = blockIdx.x;
  const int qh   = blockIdx.y;              // query half: rows [qh*128, qh*128+128)
  const int tid  = threadIdx.x;
  const int w    = tid >> 5;                // wave id 0..7
  const int lane = tid & 31;
  const int ml   = lane & 15, hs = lane >> 4;
  const int S    = nsp[0];

  const float* xg = x + b * (Cc * Ll);
  __builtin_prefetch(xg + (tid << 6), 0, 0);   // global_prefetch_b8

  // ---- stage x^T (f16) into LDS: xt[l][c], c contiguous ----
  {
    const int c = tid >> 2, lq = tid & 3;
#pragma unroll
    for (int g = 0; g < 16; ++g) {
      const int l = lq * 64 + g * 4;
      const float4 v = *(const float4*)(xg + c * Ll + l);
      smem[(l + 0) * Cc + c] = (_Float16)v.x;
      smem[(l + 1) * Cc + c] = (_Float16)v.y;
      smem[(l + 2) * Cc + c] = (_Float16)v.z;
      smem[(l + 3) * Cc + c] = (_Float16)v.w;
    }
  }
  __syncthreads();

  // ---- x A-fragments held in VGPRs: 2 K/V row tiles + 1 query row tile ----
  v16h axkv[2][2], axq[2];
#pragma unroll
  for (int mt = 0; mt < 2; ++mt)
#pragma unroll
    for (int ks = 0; ks < 2; ++ks)
      axkv[mt][ks] = load_a_frag(smem + (w * 32 + mt * 16) * Cc + ks * 32, Cc);
#pragma unroll
  for (int ks = 0; ks < 2; ++ks)
    axq[ks] = load_a_frag(smem + (qh * 128 + w * 16) * Cc + ks * 32, Cc);

  const int* sb = sidx + b * Ll;
  const int sid_kv0 = sb[w * 32 + ml];
  const int sid_kv1 = sb[w * 32 + 16 + ml];
  const int sid_q   = sb[qh * 128 + w * 16 + ml];

  // ---- weight B-fragments in VGPRs for all strokes ----
  v16h bq[2][2], bk[2][2], bwv[2][2];
#pragma unroll
  for (int nt = 0; nt < 2; ++nt)
#pragma unroll
    for (int ks = 0; ks < 2; ++ks) {
      bq[nt][ks]  = load_w_frag(wq, nt, ks);
      bk[nt][ks]  = load_w_frag(wk, nt, ks);
      bwv[nt][ks] = load_w_frag(wv, nt, ks);
    }
  const float bvv0 = bvp[ml];
  const float bvv1 = bvp[16 + ml];

  __syncthreads();   // x^T staging fully consumed -> reuse LDS for K|V

  _Float16* kh = smem;             // K: [m][32], d contiguous
  _Float16* vh = smem + Ll * Dd;   // V: [d][256], m contiguous (transposed)
  _Float16* pb = pbuf + w * (16 * 32);

  const v16h zh = {};
  v8f o_tot0 = {}, o_tot1 = {};

  for (int s = 0; s < S; ++s) {
    // ---- per-stroke masking = zeroing A-fragment rows ----
    v16h mkv[2][2], mq[2];
#pragma unroll
    for (int ks = 0; ks < 2; ++ks) {
      mkv[0][ks] = (sid_kv0 == s) ? axkv[0][ks] : zh;
      mkv[1][ks] = (sid_kv1 == s) ? axkv[1][ks] : zh;
      mq[ks]     = (sid_q   == s) ? axq[ks]     : zh;
    }

    // ---- K, V projections for this wave's 32 key rows ----
#pragma unroll
    for (int mt = 0; mt < 2; ++mt) {
      const int mb = w * 32 + mt * 16;
#pragma unroll
      for (int nt = 0; nt < 2; ++nt) {
        v8f ck = {}, cv = {};
#pragma unroll
        for (int ks = 0; ks < 2; ++ks) {
          ck = wmma16(mkv[mt][ks], bk[nt][ks], ck);
          cv = wmma16(mkv[mt][ks], bwv[nt][ks], cv);
        }
        const float bias = nt ? bvv1 : bvv0;
#pragma unroll
        for (int r = 0; r < 8; ++r) {
          const int row = mb + r + hs * 8;            // C/D layout row
          kh[row * Dd + nt * 16 + ml]   = (_Float16)ck[r];
          vh[(nt * 16 + ml) * Ll + row] = (_Float16)(cv[r] + bias);
        }
      }
    }

    // ---- Q projection for this wave's 16 query rows (transpose via pbuf) ----
#pragma unroll
    for (int nt = 0; nt < 2; ++nt) {
      v8f cq = {};
#pragma unroll
      for (int ks = 0; ks < 2; ++ks)
        cq = wmma16(mq[ks], bq[nt][ks], cq);
#pragma unroll
      for (int r = 0; r < 8; ++r)
        pb[(r + hs * 8) * Dd + nt * 16 + ml] = (_Float16)cq[r];
    }
    __syncthreads();                                  // K/V visible to all waves
    const v16h aq = load_a_frag(pb, Dd);              // Q as A-frag (K-dim = 32)

    // ---- attention: 16 queries x 256 keys, exp-sum softmax, P*V ----
    v8f os0 = {}, os1 = {};
    float rs[8];
#pragma unroll
    for (int r = 0; r < 8; ++r) rs[r] = 0.f;

#pragma unroll
    for (int j = 0; j < 8; ++j) {                     // key blocks of 32
#pragma unroll
      for (int jj = 0; jj < 2; ++jj) {
        const v16h kfrag = load_b_frag(kh + (j * 32 + jj * 16) * Dd, Dd);
        v8f ez = {};
        const v8f e = wmma16(aq, kfrag, ez);          // energy tile 16x16
#pragma unroll
        for (int r = 0; r < 8; ++r) {
          const float t = __expf(e[r] * INV_SCALE);
          rs[r] += t;
          pb[(r + hs * 8) * Dd + jj * 16 + ml] = (_Float16)t;
        }
      }
      const v16h ap = load_a_frag(pb, Dd);            // P tile 16x32 as A-frag
      {
        const v16h vf0 = load_b_frag(vh + 0 * 16 * Ll + j * 32, Ll);
        const v16h vf1 = load_b_frag(vh + 1 * 16 * Ll + j * 32, Ll);
        os0 = wmma16(ap, vf0, os0);
        os1 = wmma16(ap, vf1, os1);
      }
    }

    // row sums: reduce across the 16 lanes of each half-wave
#pragma unroll
    for (int r = 0; r < 8; ++r) {
      float t = rs[r];
      t += __shfl_xor(t, 1, 32);
      t += __shfl_xor(t, 2, 32);
      t += __shfl_xor(t, 4, 32);
      t += __shfl_xor(t, 8, 32);
      rs[r] = t;
    }
#pragma unroll
    for (int r = 0; r < 8; ++r) {
      o_tot0[r] += os0[r] / rs[r];
      o_tot1[r] += os1[r] / rs[r];
    }
    __syncthreads();          // all K/V reads done before next stroke rewrites
  }

  // ---- write out[b][d][l]; this block owns l in [qh*128, qh*128+128) ----
  float* ob = out + b * (Dd * Ll);
#pragma unroll
  for (int r = 0; r < 8; ++r) {
    const int l = qh * 128 + w * 16 + r + hs * 8;
    ob[(0 * 16 + ml) * Ll + l] = o_tot0[r];
    ob[(1 * 16 + ml) * Ll + l] = o_tot1[r];
  }
}

extern "C" void kernel_launch(void* const* d_in, const int* in_sizes, int n_in,
                              void* d_out, int out_size, void* d_ws, size_t ws_size,
                              hipStream_t stream) {
  const float* x    = (const float*)d_in[0];
  const float* wq   = (const float*)d_in[1];
  const float* wk   = (const float*)d_in[2];
  const float* wv   = (const float*)d_in[3];
  const float* bv   = (const float*)d_in[4];
  const int*   sidx = (const int*)d_in[5];
  const int*   ns   = (const int*)d_in[6];
  float* out = (float*)d_out;

  dim3 grid(64, 2);
  dim3 block(256);
  hipLaunchKernelGGL(sa_block_wmma, grid, block, 0, stream,
                     x, wq, wk, wv, bv, sidx, ns, out);
}
